// EncoderStage_86655260164512
// MI455X (gfx1250) — matrix operand order repaired and checked by reading, not compiled, not run on hardware
//
#include <hip/hip_runtime.h>
#include <hip/hip_bf16.h>

typedef __bf16 bf16_t;
typedef __attribute__((ext_vector_type(16))) __bf16 v16bf;
typedef __attribute__((ext_vector_type(8)))  float  v8f;

constexpr int B_ = 4;
constexpr int N_ = 32768;
constexpr int S_ = 8192;
constexpr int KNN_ = 16;
constexpr float EPS_ = 1e-5f;

__device__ __forceinline__ v8f zero8() {
  v8f z = {0.f,0.f,0.f,0.f,0.f,0.f,0.f,0.f};
  return z;
}

__device__ __forceinline__ v8f wmma_bf16(v16bf a, v16bf b, v8f c) {
  return __builtin_amdgcn_wmma_f32_16x16x32_bf16(false, a, false, b, (short)0, c, false, false);
}

// A fragment (16x32 bf16, M=row in LDS, K=col) per CDNA5 ISA layout:
// lane m=L%16/hi=L/16 ; VGPR v: K = (v>=4?16:0) + (v&3)*2 + 8*hi + pos
__device__ __forceinline__ v16bf load_a_frag(const bf16_t* __restrict__ s, int ld, int kt) {
  const int lane = threadIdx.x & 31;
  const int m = lane & 15;
  const int hi = lane >> 4;
  v16bf a;
#pragma unroll
  for (int i = 0; i < 16; ++i) {
    const int v = i >> 1, pos = i & 1;
    const int k = kt * 32 + ((v >= 4) ? 16 : 0) + (v & 3) * 2 + 8 * hi + pos;
    a[i] = s[m * ld + k];
  }
  return a;
}

// B fragment pre-packed lane-major: 512 bf16 per (kt,nt) tile, 16 contiguous per lane
__device__ __forceinline__ v16bf load_b_frag(const bf16_t* __restrict__ pk, int NT, int kt, int nt) {
  const int lane = threadIdx.x & 31;
  return *(const v16bf*)(pk + (((size_t)(kt * NT + nt)) * 32 + lane) * 16);
}

// ---------------- weight packer: W(O,K) row-major -> B-fragment-major bf16 ----------------
__global__ void pack_kernel(const float* __restrict__ W, bf16_t* __restrict__ dst, int Kd, int Nd) {
  int t = blockIdx.x * blockDim.x + threadIdx.x;
  int total = Kd * Nd;
  if (t >= total) return;
  int e = t & 15;
  int lane = (t >> 4) & 31;
  int f = t >> 9;
  int NT = Nd >> 4;
  int kt = f / NT, nt = f % NT;
  int hi = lane >> 4, n = lane & 15;
  int k = kt * 32 + ((e < 8) ? (8 * hi + e) : (16 + 8 * hi + (e - 8)));
  int o = nt * 16 + n;
  dst[t] = (bf16_t)W[(size_t)o * Kd + k];
}

// ---------------- q & res projections: wave handles 16 query points ----------------
__global__ __launch_bounds__(256)
void qres_kernel(const float* __restrict__ feats, const int* __restrict__ idx_fps,
                 const bf16_t* __restrict__ pkWq, const bf16_t* __restrict__ pkWres,
                 float* __restrict__ qws, float* __restrict__ resws) {
  __shared__ bf16_t sF[8][16 * 64];
  const int wave = threadIdx.x >> 5;
  const int lane = threadIdx.x & 31;
  const int ln = lane & 15, hi = lane >> 4;
  const int P0 = (blockIdx.x * 8 + wave) * 16;
  const int b = P0 / S_;
  const int row = idx_fps[P0 + ln];
  const float* frow = feats + ((size_t)b * N_ + row) * 64 + hi * 32;
#pragma unroll
  for (int j = 0; j < 32; j += 4) {
    float4 v4 = *(const float4*)(frow + j);
    bf16_t* d = &sF[wave][ln * 64 + hi * 32 + j];
    d[0] = (bf16_t)v4.x; d[1] = (bf16_t)v4.y; d[2] = (bf16_t)v4.z; d[3] = (bf16_t)v4.w;
  }
  __syncthreads();
  v16bf a0 = load_a_frag(&sF[wave][0], 64, 0);
  v16bf a1 = load_a_frag(&sF[wave][0], 64, 1);
#pragma unroll 1
  for (int nt = 0; nt < 8; ++nt) {
    v8f aq = zero8(), ar = zero8();
    aq = wmma_bf16(a0, load_b_frag(pkWq, 8, 0, nt), aq);
    aq = wmma_bf16(a1, load_b_frag(pkWq, 8, 1, nt), aq);
    ar = wmma_bf16(a0, load_b_frag(pkWres, 8, 0, nt), ar);
    ar = wmma_bf16(a1, load_b_frag(pkWres, 8, 1, nt), ar);
#pragma unroll
    for (int r = 0; r < 8; ++r) {
      size_t o = (size_t)(P0 + r + 8 * hi) * 128 + nt * 16 + ln;
      qws[o] = aq[r];
      resws[o] = ar[r];
    }
  }
}

// ---------------- attention core: wave handles one query point, M = 16 neighbors ----------------
__global__ __launch_bounds__(128)
void attn_kernel(const float* __restrict__ xyz, const float* __restrict__ feats,
                 const int* __restrict__ idx_fps, const int* __restrict__ idx_knn,
                 const float* __restrict__ Wp1,
                 const bf16_t* __restrict__ pkWk, const bf16_t* __restrict__ pkWv,
                 const bf16_t* __restrict__ pkWp2, const bf16_t* __restrict__ pkWa1,
                 const bf16_t* __restrict__ pkWa2,
                 const float* __restrict__ qws, float* __restrict__ attnws) {
  __shared__ bf16_t sNF[4][16 * 64];   // gathered n_feat (A for kk,v)
  __shared__ bf16_t sPE[4][16 * 128];  // relu(pe1) (A for pe2); later reused for relu(a1)
  __shared__ bf16_t sH[4][16 * 128];   // h = q - kk + pe (A for attn MLP)
  __shared__ bf16_t sV[4][16 * 128];   // v + pe
  const int wave = threadIdx.x >> 5;
  const int lane = threadIdx.x & 31;
  const int ln = lane & 15, hi = lane >> 4;
  const int P = blockIdx.x * 4 + wave;
  const int b = P / S_;
  const int nidx = idx_knn[(size_t)P * KNN_ + ln];
  const int cidx = idx_fps[P];
  const float* nx = xyz + ((size_t)b * N_ + nidx) * 3;
  const float* cx = xyz + ((size_t)b * N_ + cidx) * 3;
  const float r0 = nx[0] - cx[0], r1 = nx[1] - cx[1], r2 = nx[2] - cx[2];

  const float* frow = feats + ((size_t)b * N_ + nidx) * 64 + hi * 32;
#pragma unroll
  for (int j = 0; j < 32; j += 4) {
    float4 v4 = *(const float4*)(frow + j);
    bf16_t* d = &sNF[wave][ln * 64 + hi * 32 + j];
    d[0] = (bf16_t)v4.x; d[1] = (bf16_t)v4.y; d[2] = (bf16_t)v4.z; d[3] = (bf16_t)v4.w;
  }
  // pe1 = relu(rel @ Wp1^T): tiny K=3, do in VALU; lane covers 64 output channels
#pragma unroll 4
  for (int o = 0; o < 64; ++o) {
    int oc = hi * 64 + o;
    float t = r0 * Wp1[oc * 3 + 0] + r1 * Wp1[oc * 3 + 1] + r2 * Wp1[oc * 3 + 2];
    sPE[wave][ln * 128 + oc] = (bf16_t)fmaxf(t, 0.f);
  }
  __syncthreads();

  v16bf aNF0 = load_a_frag(&sNF[wave][0], 64, 0);
  v16bf aNF1 = load_a_frag(&sNF[wave][0], 64, 1);
  v16bf aPE[4];
#pragma unroll
  for (int kt = 0; kt < 4; ++kt) aPE[kt] = load_a_frag(&sPE[wave][0], 128, kt);

#pragma unroll 1
  for (int nt = 0; nt < 8; ++nt) {
    v8f zk = zero8(), zv = zero8(), zp = zero8();
    zk = wmma_bf16(aNF0, load_b_frag(pkWk, 8, 0, nt), zk);
    zk = wmma_bf16(aNF1, load_b_frag(pkWk, 8, 1, nt), zk);
    zv = wmma_bf16(aNF0, load_b_frag(pkWv, 8, 0, nt), zv);
    zv = wmma_bf16(aNF1, load_b_frag(pkWv, 8, 1, nt), zv);
#pragma unroll
    for (int kt = 0; kt < 4; ++kt)
      zp = wmma_bf16(aPE[kt], load_b_frag(pkWp2, 8, kt, nt), zp);
    float qv = qws[(size_t)P * 128 + nt * 16 + ln];
#pragma unroll
    for (int r = 0; r < 8; ++r) {
      sH[wave][(r + 8 * hi) * 128 + nt * 16 + ln] = (bf16_t)(qv - zk[r] + zp[r]);
      sV[wave][(r + 8 * hi) * 128 + nt * 16 + ln] = (bf16_t)(zv[r] + zp[r]);
    }
  }
  __syncthreads();

  // attn MLP stage 1: a1 = relu(h @ Wa1^T), N=32
  v16bf aH[4];
#pragma unroll
  for (int kt = 0; kt < 4; ++kt) aH[kt] = load_a_frag(&sH[wave][0], 128, kt);
#pragma unroll
  for (int nt2 = 0; nt2 < 2; ++nt2) {
    v8f acc = zero8();
#pragma unroll
    for (int kt = 0; kt < 4; ++kt)
      acc = wmma_bf16(aH[kt], load_b_frag(pkWa1, 2, kt, nt2), acc);
#pragma unroll
    for (int r = 0; r < 8; ++r)
      sPE[wave][(r + 8 * hi) * 128 + nt2 * 16 + ln] = (bf16_t)fmaxf(acc[r], 0.f);
  }
  __syncthreads();

  // attn MLP stage 2 + softmax over neighbors (M dim) + weighted sum of (v+pe)
  v16bf aA1 = load_a_frag(&sPE[wave][0], 128, 0);
#pragma unroll 1
  for (int nt = 0; nt < 8; ++nt) {
    v8f acc = zero8();
    acc = wmma_bf16(aA1, load_b_frag(pkWa2, 8, 0, nt), acc);
    float mx = acc[0];
#pragma unroll
    for (int r = 1; r < 8; ++r) mx = fmaxf(mx, acc[r]);
    mx = fmaxf(mx, __shfl_xor(mx, 16, 32));
    float e[8];
    float sm = 0.f;
#pragma unroll
    for (int r = 0; r < 8; ++r) { e[r] = __expf(acc[r] - mx); sm += e[r]; }
    sm += __shfl_xor(sm, 16, 32);
    float inv = 1.f / sm;
    float o = 0.f;
#pragma unroll
    for (int r = 0; r < 8; ++r)
      o += e[r] * inv * (float)sV[wave][(r + 8 * hi) * 128 + nt * 16 + ln];
    o += __shfl_xor(o, 16, 32);
    if (hi == 0) attnws[(size_t)P * 128 + nt * 16 + ln] = o;
  }
}

// ---------------- proj + residual + LayerNorm + post conv + GN partial sums ----------------
__global__ __launch_bounds__(256)
void post_kernel(const float* __restrict__ attnws, const float* __restrict__ resws,
                 const bf16_t* __restrict__ pkWproj, const bf16_t* __restrict__ pkWpost,
                 const float* __restrict__ ln_w, const float* __restrict__ ln_b,
                 float* __restrict__ yws, float* __restrict__ gnacc) {
  __shared__ bf16_t sX[8][16 * 128];
  const int wave = threadIdx.x >> 5;
  const int lane = threadIdx.x & 31;
  const int ln = lane & 15, hi = lane >> 4;
  const int P0 = (blockIdx.x * 8 + wave) * 16;
  const int b = P0 / S_;

  const float* arow = attnws + (size_t)(P0 + ln) * 128 + hi * 64;
#pragma unroll
  for (int j = 0; j < 64; j += 4) {
    float4 v4 = *(const float4*)(arow + j);
    bf16_t* d = &sX[wave][ln * 128 + hi * 64 + j];
    d[0] = (bf16_t)v4.x; d[1] = (bf16_t)v4.y; d[2] = (bf16_t)v4.z; d[3] = (bf16_t)v4.w;
  }
  __syncthreads();
  v16bf aX[4];
#pragma unroll
  for (int kt = 0; kt < 4; ++kt) aX[kt] = load_a_frag(&sX[wave][0], 128, kt);

  float t[8][8];
#pragma unroll 1
  for (int nt = 0; nt < 8; ++nt) {
    v8f acc = zero8();
#pragma unroll
    for (int kt = 0; kt < 4; ++kt)
      acc = wmma_bf16(aX[kt], load_b_frag(pkWproj, 8, kt, nt), acc);
#pragma unroll
    for (int r = 0; r < 8; ++r)
      t[nt][r] = acc[r] + resws[(size_t)(P0 + r + 8 * hi) * 128 + nt * 16 + ln];
  }
  // LayerNorm over 128 channels (8 n-tiles x 16 lanes of this half)
  float mu[8], rs[8];
#pragma unroll
  for (int r = 0; r < 8; ++r) {
    float s1 = 0.f, s2 = 0.f;
#pragma unroll
    for (int nt = 0; nt < 8; ++nt) { s1 += t[nt][r]; s2 += t[nt][r] * t[nt][r]; }
#pragma unroll
    for (int msk = 1; msk < 16; msk <<= 1) {
      s1 += __shfl_xor(s1, msk, 32);
      s2 += __shfl_xor(s2, msk, 32);
    }
    float mean = s1 * (1.f / 128.f);
    mu[r] = mean;
    rs[r] = rsqrtf(s2 * (1.f / 128.f) - mean * mean + EPS_);
  }
  __syncthreads();
#pragma unroll
  for (int nt = 0; nt < 8; ++nt)
#pragma unroll
    for (int r = 0; r < 8; ++r) {
      int c = nt * 16 + ln;
      float v = (t[nt][r] - mu[r]) * rs[r] * ln_w[c] + ln_b[c];
      sX[wave][(r + 8 * hi) * 128 + c] = (bf16_t)v;
    }
  __syncthreads();
  v16bf aY[4];
#pragma unroll
  for (int kt = 0; kt < 4; ++kt) aY[kt] = load_a_frag(&sX[wave][0], 128, kt);
#pragma unroll 1
  for (int nt = 0; nt < 8; ++nt) {
    v8f acc = zero8();
#pragma unroll
    for (int kt = 0; kt < 4; ++kt)
      acc = wmma_bf16(aY[kt], load_b_frag(pkWpost, 8, kt, nt), acc);
    float ls = 0.f, lq = 0.f;
#pragma unroll
    for (int r = 0; r < 8; ++r) {
      float yv = acc[r];
      yws[(size_t)(P0 + r + 8 * hi) * 128 + nt * 16 + ln] = yv;
      ls += yv;
      lq += yv * yv;
    }
#pragma unroll
    for (int msk = 1; msk < 32; msk <<= 1) {
      ls += __shfl_xor(ls, msk, 32);
      lq += __shfl_xor(lq, msk, 32);
    }
    if (lane == 0) {  // group == nt (16 channels per group)
      atomicAdd(&gnacc[(b * 8 + nt) * 2 + 0], ls);
      atomicAdd(&gnacc[(b * 8 + nt) * 2 + 1], lq);
    }
  }
}

// ---------------- GroupNorm finalize + ReLU ----------------
__global__ __launch_bounds__(256)
void gn_kernel(const float* __restrict__ yws, const float* __restrict__ gnacc,
               const float* __restrict__ gn_w, const float* __restrict__ gn_b,
               float* __restrict__ out) {
  int tid = blockIdx.x * blockDim.x + threadIdx.x;
  int P = tid >> 5;
  int c0 = (tid & 31) * 4;
  int b = P / S_;
  int g = c0 >> 4;
  float cnt = (float)(S_ * 16);
  float mean = gnacc[(b * 8 + g) * 2 + 0] / cnt;
  float var = gnacc[(b * 8 + g) * 2 + 1] / cnt - mean * mean;
  float inv = rsqrtf(var + EPS_);
#pragma unroll
  for (int j = 0; j < 4; ++j) {
    int c = c0 + j;
    float v = (yws[(size_t)P * 128 + c] - mean) * inv * gn_w[c] + gn_b[c];
    out[(size_t)P * 128 + c] = fmaxf(v, 0.f);
  }
}

__global__ __launch_bounds__(256)
void sxyz_kernel(const float* __restrict__ xyz, const int* __restrict__ idx_fps,
                 float* __restrict__ out) {
  int P = blockIdx.x * blockDim.x + threadIdx.x;
  if (P >= B_ * S_) return;
  int b = P / S_;
  int row = idx_fps[P];
  const float* p = xyz + ((size_t)b * N_ + row) * 3;
  out[(size_t)P * 3 + 0] = p[0];
  out[(size_t)P * 3 + 1] = p[1];
  out[(size_t)P * 3 + 2] = p[2];
}

extern "C" void kernel_launch(void* const* d_in, const int* in_sizes, int n_in,
                              void* d_out, int out_size, void* d_ws, size_t ws_size,
                              hipStream_t stream) {
  const float* xyz    = (const float*)d_in[0];
  const float* feats  = (const float*)d_in[1];
  const int* idx_fps  = (const int*)d_in[2];
  const int* idx_knn  = (const int*)d_in[3];
  const float* Wq     = (const float*)d_in[4];
  const float* Wk     = (const float*)d_in[5];
  const float* Wv     = (const float*)d_in[6];
  const float* Wp1    = (const float*)d_in[7];
  const float* Wp2    = (const float*)d_in[8];
  const float* Wa1    = (const float*)d_in[9];
  const float* Wa2    = (const float*)d_in[10];
  const float* Wproj  = (const float*)d_in[11];
  const float* ln_w   = (const float*)d_in[12];
  const float* ln_b   = (const float*)d_in[13];
  const float* Wres   = (const float*)d_in[14];
  const float* Wpost  = (const float*)d_in[15];
  const float* gn_w   = (const float*)d_in[16];
  const float* gn_b   = (const float*)d_in[17];

  char* ws = (char*)d_ws;
  size_t off = 0;
  auto take = [&](size_t bytes) -> void* {
    void* p = ws + off;
    off = (off + bytes + 255) & ~(size_t)255;
    return p;
  };
  bf16_t* pkWq    = (bf16_t*)take(64 * 128 * sizeof(bf16_t));
  bf16_t* pkWk    = (bf16_t*)take(64 * 128 * sizeof(bf16_t));
  bf16_t* pkWv    = (bf16_t*)take(64 * 128 * sizeof(bf16_t));
  bf16_t* pkWres  = (bf16_t*)take(64 * 128 * sizeof(bf16_t));
  bf16_t* pkWp2   = (bf16_t*)take(128 * 128 * sizeof(bf16_t));
  bf16_t* pkWa1   = (bf16_t*)take(128 * 32 * sizeof(bf16_t));
  bf16_t* pkWa2   = (bf16_t*)take(32 * 128 * sizeof(bf16_t));
  bf16_t* pkWproj = (bf16_t*)take(128 * 128 * sizeof(bf16_t));
  bf16_t* pkWpost = (bf16_t*)take(128 * 128 * sizeof(bf16_t));
  const size_t PTS = (size_t)B_ * S_;
  float* qws    = (float*)take(PTS * 128 * sizeof(float));
  float* resws  = (float*)take(PTS * 128 * sizeof(float));
  float* attnws = (float*)take(PTS * 128 * sizeof(float));
  float* yws    = (float*)take(PTS * 128 * sizeof(float));
  float* gnacc  = (float*)take(B_ * 8 * 2 * sizeof(float));

  hipMemsetAsync(gnacc, 0, B_ * 8 * 2 * sizeof(float), stream);

  pack_kernel<<<(64 * 128 + 255) / 256, 256, 0, stream>>>(Wq, pkWq, 64, 128);
  pack_kernel<<<(64 * 128 + 255) / 256, 256, 0, stream>>>(Wk, pkWk, 64, 128);
  pack_kernel<<<(64 * 128 + 255) / 256, 256, 0, stream>>>(Wv, pkWv, 64, 128);
  pack_kernel<<<(64 * 128 + 255) / 256, 256, 0, stream>>>(Wres, pkWres, 64, 128);
  pack_kernel<<<(128 * 128 + 255) / 256, 256, 0, stream>>>(Wp2, pkWp2, 128, 128);
  pack_kernel<<<(128 * 32 + 255) / 256, 256, 0, stream>>>(Wa1, pkWa1, 128, 32);
  pack_kernel<<<(32 * 128 + 255) / 256, 256, 0, stream>>>(Wa2, pkWa2, 32, 128);
  pack_kernel<<<(128 * 128 + 255) / 256, 256, 0, stream>>>(Wproj, pkWproj, 128, 128);
  pack_kernel<<<(128 * 128 + 255) / 256, 256, 0, stream>>>(Wpost, pkWpost, 128, 128);

  qres_kernel<<<(int)(PTS / 16 / 8), 256, 0, stream>>>(feats, idx_fps, pkWq, pkWres, qws, resws);
  attn_kernel<<<(int)(PTS / 4), 128, 0, stream>>>(xyz, feats, idx_fps, idx_knn, Wp1,
                                                  pkWk, pkWv, pkWp2, pkWa1, pkWa2, qws, attnws);
  post_kernel<<<(int)(PTS / 16 / 8), 256, 0, stream>>>(attnws, resws, pkWproj, pkWpost,
                                                       ln_w, ln_b, yws, gnacc);

  float* outf = (float*)d_out;
  sxyz_kernel<<<(int)((PTS + 255) / 256), 256, 0, stream>>>(xyz, idx_fps, outf);
  gn_kernel<<<(int)(PTS * 32 / 256), 256, 0, stream>>>(yws, gnacc, gn_w, gn_b, outf + PTS * 3);
}
